// RateRNN_27762668601491
// MI455X (gfx1250) — matrix operand order repaired and checked by reading, not compiled, not run on hardware
//
#include <hip/hip_runtime.h>

// ---------------------------------------------------------------------------
// Rate RNN (B=128, T=1000, I=16, H=512), Euler step with alpha=0.2.
// 8 persistent workgroups (16 independent batch rows each -> zero inter-WG
// sync), 1024 threads = 32 waves, one 16-wide N-tile per wave.
// GEMMs via v_wmma_f32_16x16x32_bf16 (f32 accumulate).
// W_rec/W_in feed paths per wave (K = 512 + input tile):
//   - kt 0..7 + W_in : persistent LDS panels (327KB of the WGP's 320KB LDS)
//   - kt 11..15      : pinned in VGPRs (explicit pre-loop loads, 40 regs)
//   - kt 8..10       : streamed per step from L2 via inline-asm
//                      global_load_b128 (volatile -> never hoisted)
// Steady-state pressure ~116 VGPRs: below the 128/wave cliff with headroom
// for the scheduler's ds_load pipelining.
// ---------------------------------------------------------------------------

#define RNN_B 128
#define RNN_T 1000
#define RNN_I 16
#define RNN_H 512
#define RNN_ALPHA 0.2f

#define AST 520      // tanhA row stride (ushorts): 1040B -> bank advance 4/row
#define BC_KT 8      // W_rec K-tiles cached in LDS (of 16)
#define BC_NT 9      // LDS tiles per wave: 8 W_rec + 1 W_in
#define BC_TS 536    // ushorts per LDS tile (32 lanes * 16 + swizzle pad; 1072B)
#define RT_KT 5      // W_rec K-tiles pinned in VGPRs (kt 11..15)

typedef __attribute__((ext_vector_type(16))) __bf16    v16bf;
typedef __attribute__((ext_vector_type(8)))  float     v8f;
typedef __attribute__((ext_vector_type(4)))  unsigned  v4u;

union FragB {
    v16bf v;
    v4u   q[2];
};

__device__ __forceinline__ unsigned short f32_to_bf16(float f) {
    unsigned u = __float_as_uint(f);
    unsigned r = (u + 0x7FFFu + ((u >> 16) & 1u)) >> 16;   // RNE
    return (unsigned short)r;
}

__device__ __forceinline__ float bf16_to_f32(unsigned u) {
    return __uint_as_float(u << 16);
}

// Bank-swizzled record offset (ushorts) of `lane`'s 32B B-fragment in a tile:
// plain lane*32B advances 8 banks/lane (4-way conflict); +16B per 8-lane
// group makes lanes 0..15 cover all 64 banks -> 2-way (minimum for 1KB/wave).
__device__ __forceinline__ int bc_lane_off(int lane) {
    return lane * 16 + (lane >> 3) * 8;
}

// ---------------------------------------------------------------------------
// Pack W_rec / W_in (f32 row-major [n][k]) into bf16 WMMA B-register images.
// Dense 16-bit B 32x16 layout: lane L holds column N=n0+(L&15); element e holds
// K = 8h+e (e<8) or 16+8h+(e-8) (e>=8), h=L>>4. Lane record = 32 contiguous bytes.
// pWrec: ((nt*16+kt)*32+lane)*16 + e  with n = nt*16+(lane&15), k = kt*32+kof(e)
// pWin : ( nt*32      +lane)*16 + e  with n = nt*16+(lane&15), k = kof(e) (<16 else 0)
// ---------------------------------------------------------------------------
__global__ void rnn_pack_weights(const float* __restrict__ W_in,
                                 const float* __restrict__ W_rec,
                                 unsigned short* __restrict__ pWrec,
                                 unsigned short* __restrict__ pWin) {
    int gid = blockIdx.x * 256 + threadIdx.x;
    const int NREC = 32 * 16 * 32;
    const int NIN  = 32 * 32;
    if (gid < NREC) {
        int lane = gid & 31;
        int kt   = (gid >> 5) & 15;
        int nt   = gid >> 9;
        int hh   = (lane >> 4) << 3;
        unsigned short* dst = pWrec + (size_t)gid * 16;
#pragma unroll
        for (int e = 0; e < 16; ++e)
            dst[e] = f32_to_bf16(W_rec[(size_t)(nt * 16 + (lane & 15)) * RNN_H + kt * 32 + (e < 8 ? hh + e : 16 + hh + (e - 8))]);
    } else if (gid < NREC + NIN) {
        int g    = gid - NREC;
        int lane = g & 31;
        int nt   = g >> 5;
        unsigned short* dst = pWin + (size_t)g * 16;
#pragma unroll
        for (int e = 0; e < 16; ++e)
            dst[e] = (e < 8)
                ? f32_to_bf16(W_in[(size_t)(nt * 16 + (lane & 15)) * RNN_I + ((lane >> 4) << 3) + e])
                : (unsigned short)0;
    }
}

// ---------------------------------------------------------------------------
// Persistent per-batch-tile recurrence kernel.
// LDS: 308,736 (B panels) + 16,640 (tanhA) + 512 + 1,024 + 128 = 327,040 B
//      of the WGP's 327,680 B.
// ---------------------------------------------------------------------------
__global__ __launch_bounds__(1024, 1)
void rnn_persistent_step(const float* __restrict__ inputs,
                         const unsigned short* __restrict__ pWrec,
                         const unsigned short* __restrict__ pWin,
                         const float* __restrict__ b_rec,
                         const float* __restrict__ W_out,
                         const float* __restrict__ b_out,
                         float* __restrict__ out) {
    __shared__ unsigned short bcache[32 * BC_NT * BC_TS];  // 9 tiles / wave
    __shared__ unsigned short tanhA[16 * AST];             // tanh(v) bf16
    __shared__ unsigned short xlds[16 * RNN_I];            // step inputs bf16
    __shared__ unsigned short woutb[RNN_H];                // W_out bf16
    __shared__ float yred[32];

    const int tid  = threadIdx.x;
    const int lane = tid & 31;
    const int w    = tid >> 5;              // wave id == N-tile id (0..31)
    const int n0   = w << 4;
    const int b0   = blockIdx.x << 4;       // 16 batch rows per workgroup
    const int col  = n0 + (lane & 15);      // D layout: N = lane&15 (+n0)
    const int mrow = lane & 15;             // A layout: M = lane&15
    const int moff = (lane >> 4) << 3;      // lanes>=16: K group +8 / M rows +8

    for (int h = tid; h < RNN_H; h += 1024) woutb[h] = f32_to_bf16(W_out[h]);
    for (int i = tid; i < 16 * AST; i += 1024) tanhA[i] = 0;   // tanh(0) = 0

    // Fill per-wave B panels: tiles 0..7 = W_rec kt 0..7, tile 8 = W_in.
    for (int r = tid; r < 32 * BC_NT * 32; r += 1024) {
        int nt   = r / (BC_NT * 32);
        int rem  = r - nt * (BC_NT * 32);
        int tile = rem >> 5;
        int ln   = rem & 31;
        const v4u* src = (tile < BC_KT)
            ? (const v4u*)(pWrec + ((size_t)((nt * 16 + tile) * 32 + ln) << 4))
            : (const v4u*)(pWin  + ((size_t)(nt * 32 + ln) << 4));
        v4u* dst = (v4u*)(bcache + (nt * BC_NT + tile) * BC_TS + bc_lane_off(ln));
        dst[0] = src[0];
        dst[1] = src[1];
    }

    const float breg = b_rec[col];
    const float bout = b_out[0];

    // Pin kt 11..15 in VGPRs (explicit source-level hoist; 40 regs).
    FragB creg[RT_KT];
#pragma unroll
    for (int i = 0; i < RT_KT; ++i) {
        const v4u* p = (const v4u*)(pWrec +
            ((size_t)((w * 16 + 11 + i) * 32 + lane) << 4));
        creg[i].q[0] = p[0];
        creg[i].q[1] = p[1];
    }

    v8f v = {};   // voltage, D layout: v[r] -> (m = r+moff, n = col)
    __syncthreads();

    const unsigned short* arow = tanhA + mrow * AST + moff;
    const unsigned short* brow = bcache + w * (BC_NT * BC_TS) + bc_lane_off(lane);
    const unsigned short* p8   = pWrec + ((size_t)((w * 16 + 8) * 32 + lane) << 4);

    for (int t = 0; t < RNN_T; ++t) {
        // Stage this step's inputs (f32 -> bf16) into LDS.
        if (tid < 256) {
            int m = tid >> 4, i = tid & 15;
            float xv = inputs[((size_t)(b0 + m) * RNN_T + t) * RNN_I + i];
            xlds[m * RNN_I + i] = f32_to_bf16(xv);
        }
        __syncthreads();

        // Input current: acc = x_t @ W_in^T tile (B = LDS tile 8, C = 0).
        v8f acc;
        {
            FragB xa, b;
            xa.q[0] = *(const v4u*)(xlds + mrow * RNN_I + moff);
            xa.q[1] = (v4u)(0u);                    // K 16..31 zero-pad
            b.q[0] = *(const v4u*)(brow + BC_KT * BC_TS);
            b.q[1] = *(const v4u*)(brow + BC_KT * BC_TS + 8);
            v8f czero = {};
            acc = __builtin_amdgcn_wmma_f32_16x16x32_bf16(
                false, xa.v, false, b.v, (short)0, czero, false, false);
        }

        // Recurrent GEMM: acc += tanh(v) @ W_rec^T, K = 512 in 16 tiles.
#pragma unroll
        for (int kt = 0; kt < BC_KT; ++kt) {       // LDS-resident K-tiles
            FragB a, b;
            a.q[0] = *(const v4u*)(arow + kt * 32);
            a.q[1] = *(const v4u*)(arow + kt * 32 + 16);
            b.q[0] = *(const v4u*)(brow + kt * BC_TS);
            b.q[1] = *(const v4u*)(brow + kt * BC_TS + 8);
            acc = __builtin_amdgcn_wmma_f32_16x16x32_bf16(
                false, a.v, false, b.v, (short)0, acc, false, false);
        }
        {   // kt = 8..10: streamed from L2 each step (volatile asm -> never
            // hoisted; the only per-step W_rec traffic: 3KB/wave).
            v4u q0, q1, q2, q3, q4, q5;
            asm volatile(
                "global_load_b128 %0, %6, off\n\t"
                "global_load_b128 %1, %6, off offset:16\n\t"
                "global_load_b128 %2, %6, off offset:1024\n\t"
                "global_load_b128 %3, %6, off offset:1040\n\t"
                "global_load_b128 %4, %6, off offset:2048\n\t"
                "global_load_b128 %5, %6, off offset:2064\n\t"
                "s_wait_loadcnt 0x0"
                : "=&v"(q0), "=&v"(q1), "=&v"(q2), "=&v"(q3),
                  "=&v"(q4), "=&v"(q5)
                : "v"(p8));
            FragB a, b;
            a.q[0] = *(const v4u*)(arow + 8 * 32);
            a.q[1] = *(const v4u*)(arow + 8 * 32 + 16);
            b.q[0] = q0;
            b.q[1] = q1;
            acc = __builtin_amdgcn_wmma_f32_16x16x32_bf16(
                false, a.v, false, b.v, (short)0, acc, false, false);
            a.q[0] = *(const v4u*)(arow + 9 * 32);
            a.q[1] = *(const v4u*)(arow + 9 * 32 + 16);
            b.q[0] = q2;
            b.q[1] = q3;
            acc = __builtin_amdgcn_wmma_f32_16x16x32_bf16(
                false, a.v, false, b.v, (short)0, acc, false, false);
            a.q[0] = *(const v4u*)(arow + 10 * 32);
            a.q[1] = *(const v4u*)(arow + 10 * 32 + 16);
            b.q[0] = q4;
            b.q[1] = q5;
            acc = __builtin_amdgcn_wmma_f32_16x16x32_bf16(
                false, a.v, false, b.v, (short)0, acc, false, false);
        }
#pragma unroll
        for (int kt = 11; kt < 16; ++kt) {         // VGPR-pinned K-tiles
            FragB a;
            a.q[0] = *(const v4u*)(arow + kt * 32);
            a.q[1] = *(const v4u*)(arow + kt * 32 + 16);
            acc = __builtin_amdgcn_wmma_f32_16x16x32_bf16(
                false, a.v, false, creg[kt - 11].v, (short)0, acc, false, false);
        }

        // Euler update + tanh (registers only).
        float th[8];
#pragma unroll
        for (int r = 0; r < 8; ++r) {
            float vn = (1.0f - RNN_ALPHA) * v[r] + RNN_ALPHA * (acc[r] + breg);
            v[r]  = vn;
            th[r] = tanhf(vn);
        }

        __syncthreads();   // everyone finished reading tanhA(t-1)
#pragma unroll
        for (int r = 0; r < 8; ++r)
            tanhA[(r + moff) * AST + col] = f32_to_bf16(th[r]);
        __syncthreads();   // tanhA(t) visible

        // y[b,t] = tanh(v_new[b,:]) . W_out + b_out  (wave32 shuffle reduce)
        {
            int row = w & 15;
            int hb  = ((w >> 4) << 8) + (lane << 3);
            v4u q  = *(const v4u*)(tanhA + row * AST + hb);
            v4u wq = *(const v4u*)(woutb + hb);
            float p = 0.0f;
            p += bf16_to_f32(q.x & 0xffffu) * bf16_to_f32(wq.x & 0xffffu);
            p += bf16_to_f32(q.x >> 16)     * bf16_to_f32(wq.x >> 16);
            p += bf16_to_f32(q.y & 0xffffu) * bf16_to_f32(wq.y & 0xffffu);
            p += bf16_to_f32(q.y >> 16)     * bf16_to_f32(wq.y >> 16);
            p += bf16_to_f32(q.z & 0xffffu) * bf16_to_f32(wq.z & 0xffffu);
            p += bf16_to_f32(q.z >> 16)     * bf16_to_f32(wq.z >> 16);
            p += bf16_to_f32(q.w & 0xffffu) * bf16_to_f32(wq.w & 0xffffu);
            p += bf16_to_f32(q.w >> 16)     * bf16_to_f32(wq.w >> 16);
#pragma unroll
            for (int o = 16; o >= 1; o >>= 1) p += __shfl_xor(p, o, 32);
            if (lane == 0) yred[w] = p;
        }
        __syncthreads();
        if (tid < 16) {
            out[(size_t)(b0 + tid) * RNN_T + t] = yred[tid] + yred[tid + 16] + bout;
            if (t + 1 < RNN_T)   // pull next step's input row toward the WGP
                __builtin_prefetch(
                    &inputs[((size_t)(b0 + tid) * RNN_T + (t + 1)) * RNN_I], 0, 0);
        }
    }

    // Final voltage v [B,H] appended after outputs [B,T].
#pragma unroll
    for (int r = 0; r < 8; ++r)
        out[(size_t)RNN_B * RNN_T + (size_t)(b0 + r + moff) * RNN_H + col] = v[r];
}

extern "C" void kernel_launch(void* const* d_in, const int* in_sizes, int n_in,
                              void* d_out, int out_size, void* d_ws, size_t ws_size,
                              hipStream_t stream) {
    const float* inputs = (const float*)d_in[0];   // [B,T,I]
    const float* W_in   = (const float*)d_in[1];   // [H,I]
    const float* W_rec  = (const float*)d_in[2];   // [H,H]
    const float* b_rec  = (const float*)d_in[3];   // [H]
    const float* W_out  = (const float*)d_in[4];   // [1,H]
    const float* b_out  = (const float*)d_in[5];   // [1]
    float* out = (float*)d_out;                    // [B*T] ++ [B*H]

    unsigned short* pWrec = (unsigned short*)d_ws;            // 512 KB
    unsigned short* pWin  = pWrec + (size_t)RNN_H * RNN_H;    // 32 KB

    rnn_pack_weights<<<68, 256, 0, stream>>>(W_in, W_rec, pWrec, pWin);

    rnn_persistent_step<<<RNN_B / 16, 1024, 0, stream>>>(
        inputs, pWrec, pWin, b_rec, W_out, b_out, out);
}
